// Net_28810640622216
// MI455X (gfx1250) — matrix operand order repaired and checked by reading, not compiled, not run on hardware
//
#include <hip/hip_runtime.h>
#include <hip/hip_bf16.h>

typedef __attribute__((ext_vector_type(16))) _Float16 v16h;
typedef __attribute__((ext_vector_type(8)))  _Float16 v8h;
typedef __attribute__((ext_vector_type(4)))  _Float16 v4h;
typedef __attribute__((ext_vector_type(8)))  float    v8f;

#define KS 5
#define KD 125           // 5^3 spline slots
#define NG 64            // graphs

// ---------------- degree / inverse degree ----------------
__global__ void deg_kernel(const int* __restrict__ ei, float* __restrict__ deg, int E) {
    int t = blockIdx.x * blockDim.x + threadIdx.x;
    if (t < E) atomicAdd(&deg[ei[E + t]], 1.0f);   // row 1 of edge_index = dst
}

__global__ void invdeg_kernel(float* __restrict__ deg, int n) {
    int t = blockIdx.x * blockDim.x + threadIdx.x;
    if (t < n) deg[t] = 1.0f / fmaxf(deg[t], 1.0f);
}

// ---------------- spline corner helper ----------------
__device__ __forceinline__ void spline_corner(const float p0, const float p1, const float p2,
                                              int bits, float& w, int& idx) {
    w = 1.0f; idx = 0;
    float ps[3] = {p0, p1, p2};
#pragma unroll
    for (int d = 0; d < 3; ++d) {
        float v  = ps[d] * (float)(KS - 1);
        float k0 = floorf(v);
        float fr = v - k0;
        int   b  = (bits >> d) & 1;
        w  *= b ? fr : (1.0f - fr);
        idx = idx * KS + (int)k0 + b;
    }
}

// ---------------- edge scatter, conv A (cin == 1) ----------------
__global__ void scatter_a(const int* __restrict__ ei, const float* __restrict__ pseudo,
                          const float* __restrict__ x, const float* __restrict__ invdeg,
                          float* __restrict__ acc, int E) {
    int t = blockIdx.x * blockDim.x + threadIdx.x;
    if (t >= E) return;
    int src = ei[t], dst = ei[E + t];
    float p0 = pseudo[t * 3 + 0], p1 = pseudo[t * 3 + 1], p2 = pseudo[t * 3 + 2];
    float xs = x[src];
    float inv = invdeg[dst];
#pragma unroll
    for (int bits = 0; bits < 8; ++bits) {
        float w; int idx;
        spline_corner(p0, p1, p2, bits, w, idx);
        atomicAdd(&acc[(size_t)dst * KD + idx], w * inv * xs);
    }
}

// ---------------- edge scatter, conv B (c1 channels), thread per (edge,corner) ----------------
__global__ void scatter_b(const int* __restrict__ ei, const float* __restrict__ pseudo,
                          const float* __restrict__ f1, const float* __restrict__ invdeg,
                          float* __restrict__ acc, int E, int C1) {
    int g = blockIdx.x * blockDim.x + threadIdx.x;
    if (g >= E * 8) return;
    int t = g >> 3, bits = g & 7;
    int src = ei[t], dst = ei[E + t];
    float w; int idx;
    spline_corner(pseudo[t * 3 + 0], pseudo[t * 3 + 1], pseudo[t * 3 + 2], bits, w, idx);
    float coef = w * invdeg[dst];
    size_t base = ((size_t)dst * KD + idx) * C1;
    const float* fsrc = f1 + (size_t)src * C1;
    for (int i = 0; i < C1; ++i)
        atomicAdd(&acc[base + i], coef * fsrc[i]);
}

// ---------------- LDS-staged, double-buffered WMMA GEMM ----------------
// out = relu([A|Atail] x [B;Btail] + bias), A:[M,Kmain] f32, Atail:[M,tailC],
// B:[Kmain,N], Btail:[tailC,N]; N = NTSUB*16 (compile time).
// ATOMIC=false: out[m*N+n] store.  ATOMIC=true: atomicAdd into gsum[batch[m]*64+n].
// 256 threads (8 waves) compute 64 x N; A/B staged to LDS in WMMA fragment order
// (each lane's v16h = one contiguous 32B LDS read); ping-pong buffers give one
// barrier per K-step.
template <bool ATOMIC, int NTSUB>
__global__ void spline_wmma_gemm(const float* __restrict__ A, const float* __restrict__ Atail,
                                 const float* __restrict__ B, const float* __restrict__ Btail,
                                 const float* __restrict__ bias, float* __restrict__ out,
                                 const int* __restrict__ batch,
                                 int M, int Kmain, int tailC) {
    constexpr int N     = NTSUB * 16;
    constexpr int TILES = 4 * NTSUB;     // 16x16 wmma tiles per block
    constexpr int TPW   = TILES / 8;     // tiles per wave (1 or 2)
    constexpr int EPT   = N / 8;         // B elements per staging thread (4 or 8)
    constexpr int TPR   = 32 / EPT;      // staging threads per B column

    __shared__ __align__(32) _Float16 Ash[2][64 * 32];
    __shared__ __align__(32) _Float16 Bsh[2][N * 32];

    const int tid  = threadIdx.x;
    const int lane = tid & 31;
    const int wv   = tid >> 5;
    const int half = lane >> 4;
    const int mBase = blockIdx.x * 64;
    const int Ktot  = Kmain + tailC;
    const int ksteps = (Ktot + 31) >> 5;

    // A staging: thread covers row arow, k-range [akc, akc+8); any 8-consecutive-k
    // run lands contiguously in the fragment layout.
    const int arow  = tid >> 2;                       // 0..63
    const int akc   = (tid & 3) * 8;                  // 0,8,16,24
    const int alane = (arow & 15) | (akc & 16);       // lane slot (half = bit4 of k)
    const int aj0   = akc & 15;                       // j base (k & 15)
    const int aOff  = ((((arow >> 4) << 5) | alane) << 4) | aj0;
    const int am    = mBase + arow;

    // B staging: thread covers column bn, k-range [bklc, bklc+EPT)
    const int bn    = tid / TPR;                      // 0..N-1
    const int bklc  = (tid % TPR) * EPT;
    const int blane = (bn & 15) | ((bklc >> 4) << 4);
    const int bj0   = bklc & 15;
    const int bOff  = ((((bn >> 4) << 5) | blane) << 4) | bj0;

    const bool aVec = ((Kmain & 3) == 0);

    auto stageA = [&](int kbase, _Float16* dst) {
        v8h hv;
        if (am < M && aVec && (kbase + 32) <= Kmain) {
            const float* p = A + (size_t)am * Kmain + kbase + akc;
            float4 u0 = ((const float4*)p)[0];
            float4 u1 = ((const float4*)p)[1];
            hv[0] = (_Float16)u0.x; hv[1] = (_Float16)u0.y;
            hv[2] = (_Float16)u0.z; hv[3] = (_Float16)u0.w;
            hv[4] = (_Float16)u1.x; hv[5] = (_Float16)u1.y;
            hv[6] = (_Float16)u1.z; hv[7] = (_Float16)u1.w;
            if (kbase + 64 <= Kmain) __builtin_prefetch(p + 32, 0, 1);
        } else {
#pragma unroll
            for (int i = 0; i < 8; ++i) {
                int ka = kbase + akc + i;
                float v = 0.0f;
                if (am < M) {
                    if (ka < Kmain)     v = A[(size_t)am * Kmain + ka];
                    else if (ka < Ktot) v = Atail[(size_t)am * tailC + (ka - Kmain)];
                }
                hv[i] = (_Float16)v;
            }
        }
        *(v8h*)&dst[aOff] = hv;
    };

    auto stageB = [&](int kbase, _Float16* dst) {
        _Float16 tmp[EPT];
        if ((kbase + 32) <= Kmain) {         // fast path: no guards (all but last step)
            const float* p = B + (size_t)(kbase + bklc) * N + bn;
#pragma unroll
            for (int i = 0; i < EPT; ++i) tmp[i] = (_Float16)p[(size_t)i * N];
        } else {
#pragma unroll
            for (int i = 0; i < EPT; ++i) {
                int kb = kbase + bklc + i;
                float v = 0.0f;
                if (kb < Kmain)     v = B[(size_t)kb * N + bn];
                else if (kb < Ktot) v = Btail[(size_t)(kb - Kmain) * N + bn];
                tmp[i] = (_Float16)v;
            }
        }
        if constexpr (EPT == 8) {
            v8h hb;
#pragma unroll
            for (int i = 0; i < 8; ++i) hb[i] = tmp[i];
            *(v8h*)&dst[bOff] = hb;
        } else {
            v4h hb;
#pragma unroll
            for (int i = 0; i < 4; ++i) hb[i] = tmp[i];
            *(v4h*)&dst[bOff] = hb;
        }
    };

    int msubs[TPW], nsubs[TPW];
#pragma unroll
    for (int i = 0; i < TPW; ++i) {
        int t = wv + i * 8;
        msubs[i] = t / NTSUB;
        nsubs[i] = t % NTSUB;
    }

    v8f cacc[TPW];
#pragma unroll
    for (int i = 0; i < TPW; ++i) { v8f z = {}; cacc[i] = z; }

    stageA(0, Ash[0]);
    stageB(0, Bsh[0]);
    __syncthreads();

    for (int s = 0; s < ksteps; ++s) {
        const int cur = s & 1;
        if (s + 1 < ksteps) {                 // stage next step into the other buffer
            stageA((s + 1) << 5, Ash[cur ^ 1]);
            stageB((s + 1) << 5, Bsh[cur ^ 1]);
        }
        const v16h* Av = (const v16h*)Ash[cur];
        const v16h* Bv = (const v16h*)Bsh[cur];
#pragma unroll
        for (int i = 0; i < TPW; ++i) {
            v16h a = Av[(msubs[i] << 5) | lane];
            v16h b = Bv[(nsubs[i] << 5) | lane];
            cacc[i] = __builtin_amdgcn_wmma_f32_16x16x32_f16(false, a, false, b, (short)0,
                                                             cacc[i], false, false);
        }
        __syncthreads();   // publishes next stage AND releases current buffer
    }

#pragma unroll
    for (int i = 0; i < TPW; ++i) {
        const int ncol = (nsubs[i] << 4) | (lane & 15);
        const float bv = bias[ncol];
#pragma unroll
        for (int r = 0; r < 8; ++r) {
            int m = mBase + (msubs[i] << 4) + r + (half << 3);
            if (m >= M) continue;
            float v = fmaxf(cacc[i][r] + bv, 0.0f);
            if constexpr (ATOMIC) atomicAdd(&out[(size_t)batch[m] * 64 + ncol], v);
            else                  out[(size_t)m * N + ncol] = v;
        }
    }
}

// ---------------- per-graph node counts ----------------
__global__ void gcnt_kernel(const int* __restrict__ batch, float* __restrict__ gcnt, int n) {
    int t = blockIdx.x * blockDim.x + threadIdx.x;
    if (t < n) atomicAdd(&gcnt[batch[t]], 1.0f);
}

// ---------------- final FC + log_softmax, one block per graph ----------------
__global__ void fc_logsoftmax(const float* __restrict__ gsum, const float* __restrict__ gcnt,
                              const float* __restrict__ fcw, const float* __restrict__ fcb,
                              float* __restrict__ out) {
    int b = blockIdx.x;            // graph id
    int t = threadIdx.x;           // 32 threads
    __shared__ float lg[16];
    __shared__ float mx, lse;
    if (t < 10) {
        float acc = fcb[t];
        for (int i = 0; i < 5 * 64; ++i) {
            int l = i >> 6, c = i & 63;
            float cnt = fmaxf(gcnt[l * NG + b], 1.0f);
            float h = gsum[((size_t)l * NG + b) * 64 + c] / cnt;
            acc += h * fcw[i * 10 + t];
        }
        lg[t] = acc;
    }
    __syncthreads();
    if (t == 0) {
        float m = lg[0];
        for (int j = 1; j < 10; ++j) m = fmaxf(m, lg[j]);
        float s = 0.0f;
        for (int j = 0; j < 10; ++j) s += expf(lg[j] - m);
        mx = m; lse = logf(s);
    }
    __syncthreads();
    if (t < 10) out[b * 10 + t] = lg[t] - mx - lse;
}

extern "C" void kernel_launch(void* const* d_in, const int* in_sizes, int n_in,
                              void* d_out, int out_size, void* d_ws, size_t ws_size,
                              hipStream_t stream) {
    static const int Ln[5]  = {20000, 10000, 5000, 2500, 1250};
    static const int Le[5]  = {320000, 160000, 80000, 40000, 20000};
    static const int Lc1[5] = {32, 64, 64, 64, 64};

    // ---- workspace carve-up (256B aligned); peak ~326 MB driven by acc of L1/L2
    char* ws = (char*)d_ws;
    size_t off = 0;
    auto carve = [&](size_t bytes) -> char* {
        char* p = ws + off;
        off = (off + bytes + 255) & ~(size_t)255;
        return p;
    };
    float* gsum = (float*)carve(5 * NG * 64 * sizeof(float));        // readout sums
    float* gcnt = (float*)carve(5 * NG * sizeof(float));             // node counts
    float* deg  = (float*)carve(20000 * sizeof(float));              // deg -> invdeg
    float* f1   = (float*)carve((size_t)20000 * 64 * sizeof(float)); // conv-A output
    float* acc  = (float*)carve((size_t)20000 * KD * 64 * sizeof(float)); // shared acc

    hipMemsetAsync(gsum, 0, 5 * NG * 64 * sizeof(float), stream);
    hipMemsetAsync(gcnt, 0, 5 * NG * sizeof(float), stream);

    float* outp = (float*)d_out;

    for (int l = 0; l < 5; ++l) {
        const int n = Ln[l], e = Le[l], c1 = Lc1[l];
        const int base = l * 10;
        const float* x      = (const float*)d_in[base + 0];
        const float* pseudo = (const float*)d_in[base + 1];
        const int*   ei     = (const int*)  d_in[base + 2];
        const int*   batch  = (const int*)  d_in[base + 3];
        const float* Wa     = (const float*)d_in[base + 4];   // [125,1,c1]
        const float* ra     = (const float*)d_in[base + 5];   // [1,c1]
        const float* ba     = (const float*)d_in[base + 6];
        const float* Wb     = (const float*)d_in[base + 7];   // [125,c1,64]
        const float* rb     = (const float*)d_in[base + 8];   // [c1,64]
        const float* bb     = (const float*)d_in[base + 9];

        // degree -> 1/deg
        hipMemsetAsync(deg, 0, n * sizeof(float), stream);
        deg_kernel<<<(e + 255) / 256, 256, 0, stream>>>(ei, deg, e);
        invdeg_kernel<<<(n + 255) / 256, 256, 0, stream>>>(deg, n);

        // ---- conv A: scatter (cin=1) then WMMA GEMM [n,126]x[126,c1] -> f1
        hipMemsetAsync(acc, 0, (size_t)n * KD * sizeof(float), stream);
        scatter_a<<<(e + 255) / 256, 256, 0, stream>>>(ei, pseudo, x, deg, acc, e);
        {
            int blocks = (n + 63) / 64;
            if (c1 == 32)
                spline_wmma_gemm<false, 2><<<blocks, 256, 0, stream>>>(
                    acc, x, Wa, ra, ba, f1, nullptr, n, KD, 1);
            else
                spline_wmma_gemm<false, 4><<<blocks, 256, 0, stream>>>(
                    acc, x, Wa, ra, ba, f1, nullptr, n, KD, 1);
        }

        // ---- conv B: scatter (c1 channels) then WMMA GEMM [n,126*c1]x[...,64];
        //      ReLU'd tiles atomically accumulated into per-graph sums
        hipMemsetAsync(acc, 0, (size_t)n * KD * c1 * sizeof(float), stream);
        scatter_b<<<((size_t)e * 8 + 255) / 256, 256, 0, stream>>>(ei, pseudo, f1, deg,
                                                                   acc, e, c1);
        {
            int blocks = (n + 63) / 64;
            spline_wmma_gemm<true, 4><<<blocks, 256, 0, stream>>>(
                acc, f1, Wb, rb, bb, gsum + (size_t)l * NG * 64, batch, n, KD * c1, c1);
        }
        gcnt_kernel<<<(n + 255) / 256, 256, 0, stream>>>(batch, gcnt + l * NG, n);
    }

    const float* fcw = (const float*)d_in[50];
    const float* fcb = (const float*)d_in[51];
    fc_logsoftmax<<<NG, 32, 0, stream>>>(gsum, gcnt, fcw, fcb, outp);
    (void)in_sizes; (void)n_in; (void)out_size; (void)ws_size;
}
